// LlamaSdpaAttention_26697516712268
// MI455X (gfx1250) — matrix operand order repaired and checked by reading, not compiled, not run on hardware
//
#include <hip/hip_runtime.h>

// ---------------------------------------------------------------------------
// Llama SDPA attention block for MI455X (gfx1250), bf16 WMMA implementation.
// B=2, S=2048, HIDDEN=4096, H=32 heads, G=8 kv-heads, D=128.
// ---------------------------------------------------------------------------

#define B_   2
#define S_   2048
#define H_   32
#define G_   8
#define D_   128
#define HID  4096
#define KVW  1024   // G_*D_

typedef __attribute__((ext_vector_type(16))) __bf16 bf16x16;
typedef __attribute__((ext_vector_type(8)))  float  f32x8;

union Frag {
    bf16x16      v;
    uint4        q[2];
    unsigned int u[8];
};

__device__ __forceinline__ unsigned short f2bf(float f) {
    unsigned int u = __float_as_uint(f);
    u += 0x7FFFu + ((u >> 16) & 1u);          // round-to-nearest-even
    return (unsigned short)(u >> 16);
}
__device__ __forceinline__ float bf2f(unsigned short h) {
    return __uint_as_float(((unsigned int)h) << 16);
}
__device__ __forceinline__ unsigned int packbf2(float lo, float hi) {
    return ((unsigned int)f2bf(hi) << 16) | (unsigned int)f2bf(lo);
}

// ---------------------------------------------------------------------------
// fp32 -> bf16 conversion (vectorized x4)
// ---------------------------------------------------------------------------
__global__ void cvt_f32_bf16(const float* __restrict__ in,
                             unsigned short* __restrict__ out, int n4) {
    int i = blockIdx.x * blockDim.x + threadIdx.x;
    if (i >= n4) return;
    float4 f = ((const float4*)in)[i];
    ushort4 o;
    o.x = f2bf(f.x); o.y = f2bf(f.y); o.z = f2bf(f.z); o.w = f2bf(f.w);
    ((ushort4*)out)[i] = o;
}

// ---------------------------------------------------------------------------
// RoPE on bf16 tensor laid out [token, nheads*128]. Thread d<64 updates the
// rotated pair (d, d+64). 256 threads = 4 heads per block.
// ---------------------------------------------------------------------------
__global__ void rope_kernel(unsigned short* __restrict__ T,
                            const int* __restrict__ pos, int nheads) {
    const int token = blockIdx.x;
    const int head  = blockIdx.y * 4 + (threadIdx.x >> 6);
    const int d     = threadIdx.x & 63;
    const float p        = (float)pos[token];
    // theta^(-2d/128) = exp(-ln(500000) * 2d/128)
    const float inv_freq = __expf(-13.1223633774f * ((float)(2 * d) * (1.0f / 128.0f)));
    const float fr = p * inv_freq;
    float sn, cs;
    __sincosf(fr, &sn, &cs);
    unsigned short* ptr = T + (size_t)token * ((size_t)nheads * 128) + head * 128 + d;
    const float x1 = bf2f(ptr[0]);
    const float x2 = bf2f(ptr[64]);
    ptr[0]  = f2bf(x1 * cs - x2 * sn);
    ptr[64] = f2bf(x2 * cs + x1 * sn);
}

// ---------------------------------------------------------------------------
// bf16 WMMA GEMM: C[M,N] = A[M,K] @ W[N,K]^T , fp32 accumulate.
// Block = 128 threads (4 waves, 2x2), wave tile = 64x64 (4x4 WMMA tiles).
// Fragment layouts (CDNA5 ISA 7.12.2, 16-bit, wave32):
//   A 16x32: lane(0-15)=row M, lanes 16-31 shift K by 8; elems = K[kb..kb+7],
//            K[kb+16..kb+23]  -> two 16B loads, contiguous K.
//   B 32x16: lane(0-15)=col N, lanes 16-31 shift K by 8; elems =
//            Wrow[kb..kb+7], Wrow[kb+16..kb+23] -> two 16B loads, same K order as A.
//   C 16x16 f32: lane = col N; reg r = row (hi*8 + r).
// MODE: 0 = bf16 row-major store, 1 = bf16 V^T [b,g,d,s] store, 2 = fp32 store.
// ---------------------------------------------------------------------------
template <int MODE>
__global__ __launch_bounds__(128)
void gemm_bf16(const unsigned short* __restrict__ A,
               const unsigned short* __restrict__ W,
               void* __restrict__ Cout, int M, int N, int K) {
    const int lane = threadIdx.x & 31;
    const int wid  = threadIdx.x >> 5;
    const int lo   = lane & 15;
    const int hi   = lane >> 4;
    const int rowBase = blockIdx.y * 128 + (wid >> 1) * 64;
    const int colBase = blockIdx.x * 128 + (wid & 1) * 64;

    const f32x8 zero = {0.f, 0.f, 0.f, 0.f, 0.f, 0.f, 0.f, 0.f};
    f32x8 acc[4][4];
    #pragma unroll
    for (int i = 0; i < 4; i++)
        #pragma unroll
        for (int j = 0; j < 4; j++) acc[i][j] = zero;

    const unsigned short* Arow[4];
    const unsigned short* Wrow[4];
    #pragma unroll
    for (int i = 0; i < 4; i++) Arow[i] = A + (size_t)(rowBase + i * 16 + lo) * K;
    #pragma unroll
    for (int j = 0; j < 4; j++) Wrow[j] = W + (size_t)(colBase + j * 16 + lo) * K;

    for (int k0 = 0; k0 < K; k0 += 32) {
        const int ka = k0 + hi * 8;
        const int kb = k0 + hi * 8;
        Frag a[4], b[4];
        #pragma unroll
        for (int i = 0; i < 4; i++) {
            a[i].q[0] = *(const uint4*)(Arow[i] + ka);
            a[i].q[1] = *(const uint4*)(Arow[i] + ka + 16);
        }
        #pragma unroll
        for (int j = 0; j < 4; j++) {
            b[j].q[0] = *(const uint4*)(Wrow[j] + kb);
            b[j].q[1] = *(const uint4*)(Wrow[j] + kb + 16);
        }
        if (k0 + 32 < K) {  // lowers to global_prefetch_b8 on gfx1250
            __builtin_prefetch(Arow[0] + ka + 32, 0, 1);
            __builtin_prefetch(Wrow[0] + kb + 32, 0, 1);
        }
        #pragma unroll
        for (int i = 0; i < 4; i++)
            #pragma unroll
            for (int j = 0; j < 4; j++)
                acc[i][j] = __builtin_amdgcn_wmma_f32_16x16x32_bf16(
                    false, a[i].v, false, b[j].v, (short)0, acc[i][j], false, false);
    }

    if (MODE == 1) {
        // V^T store: [ (b*G+g)*D + d ][ s ] with s contiguous per lane.
        unsigned short* C = (unsigned short*)Cout;
        #pragma unroll
        for (int i = 0; i < 4; i++)
            #pragma unroll
            for (int j = 0; j < 4; j++) {
                const int n  = colBase + j * 16 + lo;   // g*128 + d
                const int g  = n >> 7, d = n & 127;
                const int mb = rowBase + i * 16 + hi * 8;  // token base
                const int bb = mb >> 11, s = mb & 2047;
                unsigned short* p =
                    C + (((size_t)(bb * G_ + g) * D_ + d) * S_ + s);
                uint4 pk;
                pk.x = packbf2(acc[i][j][0], acc[i][j][1]);
                pk.y = packbf2(acc[i][j][2], acc[i][j][3]);
                pk.z = packbf2(acc[i][j][4], acc[i][j][5]);
                pk.w = packbf2(acc[i][j][6], acc[i][j][7]);
                *(uint4*)p = pk;
            }
    } else if (MODE == 2) {
        float* C = (float*)Cout;
        #pragma unroll
        for (int i = 0; i < 4; i++)
            #pragma unroll
            for (int j = 0; j < 4; j++) {
                const int n  = colBase + j * 16 + lo;
                const int mb = rowBase + i * 16 + hi * 8;
                #pragma unroll
                for (int r = 0; r < 8; r++)
                    C[(size_t)(mb + r) * N + n] = acc[i][j][r];
            }
    } else {
        unsigned short* C = (unsigned short*)Cout;
        #pragma unroll
        for (int i = 0; i < 4; i++)
            #pragma unroll
            for (int j = 0; j < 4; j++) {
                const int n  = colBase + j * 16 + lo;
                const int mb = rowBase + i * 16 + hi * 8;
                #pragma unroll
                for (int r = 0; r < 8; r++)
                    C[(size_t)(mb + r) * N + n] = f2bf(acc[i][j][r]);
            }
    }
}

// ---------------------------------------------------------------------------
// Flash attention (causal, GQA). One wave = 16 queries of one (b,h).
// S^T = K·Q^T is computed so that each lane owns a single query column:
// softmax reductions are 16 in-lane values + one __shfl_xor(16) (wave32).
// PV is acc^T += V^T · P^T; the P B-fragment is assembled with two xor-16
// shuffles. V^T was pre-stored by the V-projection GEMM epilogue.
// ---------------------------------------------------------------------------
__global__ __launch_bounds__(128)
void flash_attn(const unsigned short* __restrict__ Qb,
                const unsigned short* __restrict__ Kb,
                const unsigned short* __restrict__ VTb,
                unsigned short* __restrict__ attnb) {
    const int lane = threadIdx.x & 31;
    const int wid  = threadIdx.x >> 5;
    const int lo   = lane & 15;
    const int hi   = lane >> 4;
    const int b = blockIdx.z, h = blockIdx.y, g = h >> 2;  // rep = H/G = 4
    const int qwave = blockIdx.x * 64 + wid * 16;
    const float scale = 0.08838834764831845f;  // 1/sqrt(128)

    // Preload Q as B-operand fragments of Q^T (lane = query column; each lane
    // reads 16 contiguous bf16 of its own Q row per 32-d chunk).
    const unsigned short* qptr = Qb + (size_t)(b * S_ + qwave + lo) * HID + h * D_;
    Frag qf[4];
    #pragma unroll
    for (int c = 0; c < 4; c++) {
        qf[c].q[0] = *(const uint4*)(qptr + c * 32 + hi * 16);
        qf[c].q[1] = *(const uint4*)(qptr + c * 32 + hi * 16 + 8);
    }

    const f32x8 zero = {0.f, 0.f, 0.f, 0.f, 0.f, 0.f, 0.f, 0.f};
    f32x8 accT[8];
    #pragma unroll
    for (int t = 0; t < 8; t++) accT[t] = zero;
    float m_i = -3.0e38f, l_i = 0.0f;

    const int qmax = qwave + 15;
    for (int k0 = 0; k0 <= qmax; k0 += 32) {
        float sc[2][8];
        #pragma unroll
        for (int ks = 0; ks < 2; ks++) {
            const int kb_abs = k0 + ks * 16;
            const unsigned short* kp =
                Kb + (size_t)(b * S_ + kb_abs + lo) * KVW + g * D_;
            f32x8 st = zero;
            #pragma unroll
            for (int c = 0; c < 4; c++) {
                Frag a;  // K tile as A-operand (lane = key row, contiguous d)
                a.q[0] = *(const uint4*)(kp + c * 32 + hi * 16);
                a.q[1] = *(const uint4*)(kp + c * 32 + hi * 16 + 8);
                st = __builtin_amdgcn_wmma_f32_16x16x32_bf16(
                    false, a.v, false, qf[c].v, (short)0, st, false, false);
            }
            #pragma unroll
            for (int r = 0; r < 8; r++) {
                const int kabs = kb_abs + hi * 8 + r;  // row m of S^T = key
                const float v = st[r] * scale;
                sc[ks][r] = (kabs > (qwave + lo)) ? -3.0e38f : v;
            }
        }
        // ---- online softmax (per query = per lane column) ----
        float mt = sc[0][0];
        #pragma unroll
        for (int r = 1; r < 8; r++) mt = fmaxf(mt, sc[0][r]);
        #pragma unroll
        for (int r = 0; r < 8; r++) mt = fmaxf(mt, sc[1][r]);
        mt = fmaxf(mt, __shfl_xor(mt, 16, 32));
        const float m_new = fmaxf(m_i, mt);
        const float alpha = __expf(m_i - m_new);
        float p0[8], p1[8], lsum = 0.0f;
        #pragma unroll
        for (int r = 0; r < 8; r++) { p0[r] = __expf(sc[0][r] - m_new); lsum += p0[r]; }
        #pragma unroll
        for (int r = 0; r < 8; r++) { p1[r] = __expf(sc[1][r] - m_new); lsum += p1[r]; }
        lsum += __shfl_xor(lsum, 16, 32);
        l_i = l_i * alpha + lsum;
        m_i = m_new;
        #pragma unroll
        for (int t = 0; t < 8; t++)
            #pragma unroll
            for (int r = 0; r < 8; r++) accT[t][r] *= alpha;

        // ---- pack P^T into a 32x16 B-fragment (keys x queries) ----
        unsigned int pk0[4], pk1[4];
        #pragma unroll
        for (int w = 0; w < 4; w++) {
            pk0[w] = packbf2(p0[2 * w], p0[2 * w + 1]);
            pk1[w] = packbf2(p1[2 * w], p1[2 * w + 1]);
        }
        Frag bp;
        #pragma unroll
        for (int w = 0; w < 4; w++) {
            const unsigned int o0 = (unsigned int)__shfl_xor((int)pk0[w], 16, 32);
            const unsigned int o1 = (unsigned int)__shfl_xor((int)pk1[w], 16, 32);
            bp.u[w]     = hi ? o1     : pk0[w];  // keys (hi?16:0)+0..7
            bp.u[w + 4] = hi ? pk1[w] : o0;      // keys (hi?24:8)+0..7
        }
        // ---- acc^T += V^T(16d x 32k) · P^T(32k x 16q) ----
        #pragma unroll
        for (int t = 0; t < 8; t++) {
            const unsigned short* vp =
                VTb + ((size_t)(b * G_ + g) * D_ + t * 16 + lo) * S_ + k0;
            Frag av;
            av.q[0] = *(const uint4*)(vp + hi * 16);
            av.q[1] = *(const uint4*)(vp + hi * 16 + 8);
            accT[t] = __builtin_amdgcn_wmma_f32_16x16x32_bf16(
                false, av.v, false, bp.v, (short)0, accT[t], false, false);
        }
    }

    // ---- epilogue: attn[token, h*128+d] = acc^T / l  (lane = its query) ----
    const float inv_l = 1.0f / l_i;
    unsigned short* op = attnb + (size_t)(b * S_ + qwave + lo) * HID + h * D_;
    #pragma unroll
    for (int t = 0; t < 8; t++) {
        uint4 pk;
        pk.x = packbf2(accT[t][0] * inv_l, accT[t][1] * inv_l);
        pk.y = packbf2(accT[t][2] * inv_l, accT[t][3] * inv_l);
        pk.z = packbf2(accT[t][4] * inv_l, accT[t][5] * inv_l);
        pk.w = packbf2(accT[t][6] * inv_l, accT[t][7] * inv_l);
        *(uint4*)(op + t * 16 + hi * 8) = pk;
    }
}

// ---------------------------------------------------------------------------
// Host launcher
// ---------------------------------------------------------------------------
extern "C" void kernel_launch(void* const* d_in, const int* in_sizes, int n_in,
                              void* d_out, int out_size, void* d_ws, size_t ws_size,
                              hipStream_t stream) {
    const float* x  = (const float*)d_in[0];
    const int*   pos = (const int*)d_in[1];
    const float* wq = (const float*)d_in[2];
    const float* wk = (const float*)d_in[3];
    const float* wv = (const float*)d_in[4];
    const float* wo = (const float*)d_in[5];
    float* out = (float*)d_out;

    const size_t NX  = (size_t)B_ * S_ * HID;   // 16,777,216
    const size_t NWQ = (size_t)HID * HID;       // 16,777,216
    const size_t NWK = (size_t)KVW * HID;       //  4,194,304
    const size_t NKV = (size_t)B_ * S_ * KVW;   //  4,194,304

    unsigned short* p    = (unsigned short*)d_ws;
    unsigned short* xb   = p; p += NX;
    unsigned short* wqb  = p; p += NWQ;
    unsigned short* wkb  = p; p += NWK;
    unsigned short* wvb  = p; p += NWK;
    unsigned short* wob  = p; p += NWQ;
    unsigned short* Qb   = p; p += NX;
    unsigned short* Kb   = p; p += NKV;
    unsigned short* VTb  = p; p += NKV;
    unsigned short* attnb = p; p += NX;

    auto cvt = [&](const float* src, unsigned short* dst, size_t n) {
        const int n4 = (int)(n / 4);
        cvt_f32_bf16<<<dim3((n4 + 255) / 256), dim3(256), 0, stream>>>(src, dst, n4);
    };
    cvt(x,  xb,  NX);
    cvt(wq, wqb, NWQ);
    cvt(wk, wkb, NWK);
    cvt(wv, wvb, NWK);
    cvt(wo, wob, NWQ);

    const int M = B_ * S_;  // 4096 tokens
    gemm_bf16<0><<<dim3(HID / 128, M / 128), 128, 0, stream>>>(xb, wqb, Qb,  M, HID, HID);
    gemm_bf16<0><<<dim3(KVW / 128, M / 128), 128, 0, stream>>>(xb, wkb, Kb,  M, KVW, HID);
    gemm_bf16<1><<<dim3(KVW / 128, M / 128), 128, 0, stream>>>(xb, wvb, VTb, M, KVW, HID);

    rope_kernel<<<dim3(M, H_ / 4), 256, 0, stream>>>(Qb, pos, H_);
    rope_kernel<<<dim3(M, G_ / 4), 256, 0, stream>>>(Kb, pos, G_);

    flash_attn<<<dim3(S_ / 64, H_, B_), 128, 0, stream>>>(Qb, Kb, VTb, attnb);

    gemm_bf16<2><<<dim3(HID / 128, M / 128), 128, 0, stream>>>(attnb, wob, out, M, HID, HID);
}
